// Attention_28278064677229
// MI455X (gfx1250) — matrix operand order repaired and checked by reading, not compiled, not run on hardware
//
#include <hip/hip_runtime.h>

typedef __attribute__((ext_vector_type(16))) __bf16 v16bf;
typedef __attribute__((ext_vector_type(8)))  __bf16 v8bf;
typedef __attribute__((ext_vector_type(8)))  float  v8f;
typedef __attribute__((ext_vector_type(4)))  int    v4i;

#define DEV static __device__ __forceinline__

#if defined(__AMDGCN__) && \
    __has_builtin(__builtin_amdgcn_global_load_async_to_lds_b128) && \
    __has_builtin(__builtin_amdgcn_s_wait_asynccnt)
#define HAVE_ASYNC_LDS 1
#else
#define HAVE_ASYNC_LDS 0
#endif

// --------------------------------------------------------------------------
// WMMA helper: D = A(16x32 bf16) * B(32x16 bf16) + C(16x16 f32)
// --------------------------------------------------------------------------
DEV v8f wmma_bf16(v16bf a, v16bf b, v8f c) {
  return __builtin_amdgcn_wmma_f32_16x16x32_bf16(
      /*neg_a=*/false, a, /*neg_b=*/false, b,
      /*c_mod=*/(short)0, c, /*reuse_a=*/false, /*reuse_b=*/false);
}

// A-fragment (16x32, row-major source, stride ld elements).
// ISA layout: lane half h=lane>>4, row M=lane&15;
//   elements 0..7  = K in [h*8, h*8+7], elements 8..15 = K in [16+h*8, +7]
DEV v16bf load_a_frag(const __bf16* base, int ld, int lane) {
  const int r = lane & 15;
  const int h = lane >> 4;
  const v8bf lo = *(const v8bf*)(base + (size_t)r * ld + h * 8);
  const v8bf hi = *(const v8bf*)(base + (size_t)r * ld + 16 + h * 8);
  v16bf a;
#pragma unroll
  for (int i = 0; i < 8; ++i) { a[i] = lo[i]; a[8 + i] = hi[i]; }
  return a;
}

// B-fragment (32x16 KxN, row-major source): N = lane&15,
//   elements 0..7 = K in [h*8, h*8+7], elements 8..15 = K in [16+h*8, +7]
DEV v16bf load_b_frag(const __bf16* base, int ld, int lane) {
  const int c = lane & 15;
  const int h = lane >> 4;
  v16bf b;
#pragma unroll
  for (int i = 0; i < 8; ++i) {
    b[i] = base[(size_t)(h * 8 + i) * ld + c];
    b[8 + i] = base[(size_t)(16 + h * 8 + i) * ld + c];
  }
  return b;
}

// 16-byte global -> LDS copy, async (ASYNCcnt) when available.
DEV void copy16_to_lds(const __bf16* g, __bf16* l) {
#if HAVE_ASYNC_LDS
  __builtin_amdgcn_global_load_async_to_lds_b128(
      (__attribute__((address_space(1))) v4i*)
          (__attribute__((address_space(1))) const void*)g,
      (__attribute__((address_space(3))) v4i*)
          (__attribute__((address_space(3))) void*)l,
      /*offset=*/0, /*cpol=*/0);
#else
  *(v8bf*)l = *(const v8bf*)g;
#endif
}

template <int N> DEV void wait_async() {
#if HAVE_ASYNC_LDS
  __builtin_amdgcn_s_wait_asynccnt(N);
#endif
}

// --------------------------------------------------------------------------
// fp32 -> bf16 conversion
// --------------------------------------------------------------------------
__global__ __launch_bounds__(256) void cvt_bf16_kernel(
    const float* __restrict__ in, __bf16* __restrict__ out, int n) {
  int i = blockIdx.x * 256 + threadIdx.x;
  if (i < n) out[i] = (__bf16)in[i];
}

// --------------------------------------------------------------------------
// QKV GEMM: X[8192x768] * Wqkv[768x2304] -> Q/V as [B,H,N,96] bf16,
// K transposed as [B,H,96,N] bf16.   4 waves/block, 64x64 block tile.
// --------------------------------------------------------------------------
__global__ __launch_bounds__(128) void qkv_gemm_kernel(
    const __bf16* __restrict__ X, const __bf16* __restrict__ W,
    __bf16* __restrict__ Q, __bf16* __restrict__ Kt, __bf16* __restrict__ V) {
  const int lane = threadIdx.x & 31;
  const int wave = threadIdx.x >> 5;
  const int rowBase = blockIdx.y * 64 + (wave >> 1) * 32;
  const int colBase = blockIdx.x * 64 + (wave & 1) * 32;

  v8f acc[2][2] = {};
  for (int kk = 0; kk < 768; kk += 32) {
    v16bf a0 = load_a_frag(X + (size_t)rowBase * 768 + kk, 768, lane);
    v16bf a1 = load_a_frag(X + (size_t)(rowBase + 16) * 768 + kk, 768, lane);
    v16bf b0 = load_b_frag(W + (size_t)kk * 2304 + colBase, 2304, lane);
    v16bf b1 = load_b_frag(W + (size_t)kk * 2304 + colBase + 16, 2304, lane);
    acc[0][0] = wmma_bf16(a0, b0, acc[0][0]);
    acc[0][1] = wmma_bf16(a0, b1, acc[0][1]);
    acc[1][0] = wmma_bf16(a1, b0, acc[1][0]);
    acc[1][1] = wmma_bf16(a1, b1, acc[1][1]);
  }

  const int colOff = lane & 15;
  const int rowOff = (lane >> 4) * 8;
#pragma unroll
  for (int ti = 0; ti < 2; ++ti)
#pragma unroll
    for (int tj = 0; tj < 2; ++tj)
#pragma unroll
      for (int i = 0; i < 8; ++i) {
        const int m = rowBase + ti * 16 + rowOff + i;   // 0..8191
        const int n = colBase + tj * 16 + colOff;       // 0..2303
        const int s   = n / 768;                        // 0=Q 1=K 2=V
        const int rem = n - s * 768;
        const int h   = rem / 96;
        const int d   = rem - h * 96;
        const int b   = m >> 12;
        const int row = m & 4095;
        const int bh  = b * 8 + h;
        const __bf16 val = (__bf16)acc[ti][tj][i];
        if (s == 0)      Q [(size_t)(bh * 4096 + row) * 96 + d]   = val;
        else if (s == 1) Kt[(size_t)(bh * 96 + d) * 4096 + row]   = val;
        else             V [(size_t)(bh * 4096 + row) * 96 + d]   = val;
      }
}

// --------------------------------------------------------------------------
// Flash attention. Workgroup = 4 waves = 64 queries of one (b,h).
// K/V tiles for each 32-key block are staged in LDS (async, double-buffered)
// and shared by all 4 waves. Per wave: S = Q*K^T (6 WMMA), online softmax,
// O += P*V (6 WMMA). Output bf16 [B,N,C] with heads re-interleaved.
// --------------------------------------------------------------------------
__global__ __launch_bounds__(128) void attn_kernel(
    const __bf16* __restrict__ Q, const __bf16* __restrict__ Kt,
    const __bf16* __restrict__ V, __bf16* __restrict__ O) {
  __shared__ __bf16 kbuf[2][96 * 32];   // [d][key]   6 KB each
  __shared__ __bf16 vbuf[2][32 * 96];   // [key][d]   6 KB each
  __shared__ __bf16 pbuf[4][16 * 32];   // per-wave P staging

  const int lane = threadIdx.x & 31;
  const int wave = threadIdx.x >> 5;
  const int tid  = threadIdx.x;

  const int bh   = blockIdx.x >> 6;            // 0..15
  const int qblk = blockIdx.x & 63;            // 0..63
  const int q0   = qblk * 64 + wave * 16;
  const int h    = bh & 7;
  const int b    = bh >> 3;

  const __bf16* Qp  = Q  + (size_t)(bh * 4096 + q0) * 96;
  const __bf16* Ktp = Kt + (size_t)bh * 96 * 4096;
  const __bf16* Vp  = V  + (size_t)bh * 4096 * 96;
  const float scale = 0.1020620726159658f;     // 96^-0.5

  v16bf aq[3];
#pragma unroll
  for (int ds = 0; ds < 3; ++ds) aq[ds] = load_a_frag(Qp + ds * 32, 96, lane);

  v8f o_acc[6] = {};
  float m_run[8], l_run[8];
#pragma unroll
  for (int i = 0; i < 8; ++i) { m_run[i] = -3.0e38f; l_run[i] = 0.0f; }

  __bf16* pb = pbuf[wave];
  const int colb   = lane & 15;
  const int rowOff = (lane >> 4) * 8;

  // Stage one 32-key block: 6 KB K-tile + 6 KB V-tile, 6 x b128 per thread.
  auto stage_kv = [&](int key0, int sel) {
    __bf16* kb = kbuf[sel];
    __bf16* vb = vbuf[sel];
#pragma unroll
    for (int it = 0; it < 3; ++it) {            // K-tile: 384 16B chunks
      const int c = it * 128 + tid;
      const int d = c >> 2, q = (c & 3) * 8;
      copy16_to_lds(Ktp + (size_t)d * 4096 + key0 + q, kb + d * 32 + q);
    }
#pragma unroll
    for (int it = 0; it < 3; ++it) {            // V-tile: 384 16B chunks
      const int c = it * 128 + tid;
      const int key = c / 12, j = (c % 12) * 8;
      copy16_to_lds(Vp + (size_t)(key0 + key) * 96 + j, vb + key * 96 + j);
    }
  };

  stage_kv(0, 0);                               // prologue: block 0 -> buf 0

  for (int kb = 0; kb < 128; ++kb) {
    const int sel = kb & 1;
    if (kb + 1 < 128) {
      stage_kv((kb + 1) * 32, sel ^ 1);         // prefetch next block
      wait_async<6>();                          // oldest group (this block) done
    } else {
      wait_async<0>();
    }
    __syncthreads();                            // publish staged K/V

    const __bf16* kt_l = kbuf[sel];
    const __bf16* v_l  = vbuf[sel];

    v8f s0 = {}, s1 = {};
#pragma unroll
    for (int ds = 0; ds < 3; ++ds) {
      s0 = wmma_bf16(aq[ds], load_b_frag(kt_l + (ds * 32) * 32,      32, lane), s0);
      s1 = wmma_bf16(aq[ds], load_b_frag(kt_l + (ds * 32) * 32 + 16, 32, lane), s1);
    }

    // online softmax; f32-frag element i = row rowOff+i, cols across 16 lanes
    float corr[8];
#pragma unroll
    for (int i = 0; i < 8; ++i) {
      const float sa = s0[i] * scale;
      const float sb = s1[i] * scale;
      s0[i] = sa; s1[i] = sb;
      float v = fmaxf(sa, sb);
#pragma unroll
      for (int off = 1; off < 16; off <<= 1) v = fmaxf(v, __shfl_xor(v, off, 32));
      const float nm = fmaxf(m_run[i], v);
      corr[i] = __expf(m_run[i] - nm);
      m_run[i] = nm;
    }
#pragma unroll
    for (int i = 0; i < 8; ++i) {
      const float pa = __expf(s0[i] - m_run[i]);
      const float pc = __expf(s1[i] - m_run[i]);
      s0[i] = pa; s1[i] = pc;
      float r = pa + pc;
#pragma unroll
      for (int off = 1; off < 16; off <<= 1) r += __shfl_xor(r, off, 32);
      l_run[i] = l_run[i] * corr[i] + r;
    }
#pragma unroll
    for (int dt = 0; dt < 6; ++dt)
#pragma unroll
      for (int i = 0; i < 8; ++i) o_acc[dt][i] *= corr[i];

    // P: C-layout regs -> LDS row-major 16x32 -> reload as A-fragment.
#pragma unroll
    for (int i = 0; i < 8; ++i) {
      pb[(rowOff + i) * 32 + colb]      = (__bf16)s0[i];
      pb[(rowOff + i) * 32 + 16 + colb] = (__bf16)s1[i];
    }
    const v16bf ap = load_a_frag(pb, 32, lane); // wave-private, DS in-order

#pragma unroll
    for (int dt = 0; dt < 6; ++dt)
      o_acc[dt] = wmma_bf16(ap, load_b_frag(v_l + dt * 16, 96, lane), o_acc[dt]);

    __syncthreads();                            // everyone done with buf[sel]
  }

  float inv[8];
#pragma unroll
  for (int i = 0; i < 8; ++i) inv[i] = 1.0f / l_run[i];
#pragma unroll
  for (int dt = 0; dt < 6; ++dt)
#pragma unroll
    for (int i = 0; i < 8; ++i) {
      const int row = b * 4096 + q0 + rowOff + i;
      const int col = h * 96 + dt * 16 + colb;
      O[(size_t)row * 768 + col] = (__bf16)(o_acc[dt][i] * inv[i]);
    }
}

// --------------------------------------------------------------------------
// Projection GEMM: O[8192x768] * Wproj[768x768] + bias -> out f32
// --------------------------------------------------------------------------
__global__ __launch_bounds__(128) void proj_gemm_kernel(
    const __bf16* __restrict__ A, const __bf16* __restrict__ W,
    const float* __restrict__ bias, float* __restrict__ out) {
  const int lane = threadIdx.x & 31;
  const int wave = threadIdx.x >> 5;
  const int rowBase = blockIdx.y * 64 + (wave >> 1) * 32;
  const int colBase = blockIdx.x * 64 + (wave & 1) * 32;

  v8f acc[2][2] = {};
  for (int kk = 0; kk < 768; kk += 32) {
    v16bf a0 = load_a_frag(A + (size_t)rowBase * 768 + kk, 768, lane);
    v16bf a1 = load_a_frag(A + (size_t)(rowBase + 16) * 768 + kk, 768, lane);
    v16bf b0 = load_b_frag(W + (size_t)kk * 768 + colBase, 768, lane);
    v16bf b1 = load_b_frag(W + (size_t)kk * 768 + colBase + 16, 768, lane);
    acc[0][0] = wmma_bf16(a0, b0, acc[0][0]);
    acc[0][1] = wmma_bf16(a0, b1, acc[0][1]);
    acc[1][0] = wmma_bf16(a1, b0, acc[1][0]);
    acc[1][1] = wmma_bf16(a1, b1, acc[1][1]);
  }

  const int colOff = lane & 15;
  const int rowOff = (lane >> 4) * 8;
#pragma unroll
  for (int ti = 0; ti < 2; ++ti)
#pragma unroll
    for (int tj = 0; tj < 2; ++tj) {
      const int n = colBase + tj * 16 + colOff;
      const float bv = bias[n];
#pragma unroll
      for (int i = 0; i < 8; ++i) {
        const int m = rowBase + ti * 16 + rowOff + i;
        out[(size_t)m * 768 + n] = acc[ti][tj][i] + bv;
      }
    }
}

// --------------------------------------------------------------------------
// Host launcher
// --------------------------------------------------------------------------
extern "C" void kernel_launch(void* const* d_in, const int* in_sizes, int n_in,
                              void* d_out, int out_size, void* d_ws, size_t ws_size,
                              hipStream_t stream) {
  (void)in_sizes; (void)n_in; (void)out_size; (void)ws_size;
  const float* x      = (const float*)d_in[0];  // [2,4096,768]
  const float* w_qkv  = (const float*)d_in[1];  // [768,2304]
  const float* w_proj = (const float*)d_in[2];  // [768,768]
  const float* b_proj = (const float*)d_in[3];  // [768]
  float* out = (float*)d_out;                   // [2,4096,768] f32

  const int BN = 2 * 4096;          // 8192
  const int C  = 768;
  const int C3 = 2304;
  const size_t nX  = (size_t)BN * C;
  const size_t nWq = (size_t)C * C3;
  const size_t nWp = (size_t)C * C;

  char* ws = (char*)d_ws;
  size_t off = 0;
  auto take = [&](size_t bytes) { size_t r = off; off += (bytes + 255) & ~(size_t)255; return r; };
  __bf16* xb   = (__bf16*)(ws + take(nX  * 2));
  __bf16* wqb  = (__bf16*)(ws + take(nWq * 2));
  __bf16* wpb  = (__bf16*)(ws + take(nWp * 2));
  __bf16* Qb   = (__bf16*)(ws + take(nX  * 2));  // [B,H,N,96]
  __bf16* Ktb  = (__bf16*)(ws + take(nX  * 2));  // [B,H,96,N]
  __bf16* Vb   = (__bf16*)(ws + take(nX  * 2));  // [B,H,N,96]
  __bf16* Ob   = (__bf16*)(ws + take(nX  * 2));  // [B,N,C]

  cvt_bf16_kernel<<<(nX  + 255) / 256, 256, 0, stream>>>(x,      xb,  (int)nX);
  cvt_bf16_kernel<<<(nWq + 255) / 256, 256, 0, stream>>>(w_qkv,  wqb, (int)nWq);
  cvt_bf16_kernel<<<(nWp + 255) / 256, 256, 0, stream>>>(w_proj, wpb, (int)nWp);

  qkv_gemm_kernel<<<dim3(C3 / 64, BN / 64), 128, 0, stream>>>(xb, wqb, Qb, Ktb, Vb);

  // 16 (b,h) x 64 query-blocks of 64 queries = 1024 workgroups
  attn_kernel<<<1024, 128, 0, stream>>>(Qb, Ktb, Vb, Ob);

  proj_gemm_kernel<<<dim3(C / 64, BN / 64), 128, 0, stream>>>(Ob, wpb, b_proj, out);
}
